// DGCNNCls_712964571700
// MI455X (gfx1250) — matrix operand order repaired and checked by reading, not compiled, not run on hardware
//
#include <hip/hip_runtime.h>
#include <hip/hip_bf16.h>

// ---------------------------------------------------------------------------
// DGCNN classifier forward for MI455X (gfx1250), wave32 + WMMA f16 16x16x32.
//
// Math restructuring (exact up to fp reassociation):
//   edge_conv:  h[b,n,j,o] = t1[b,idx,o] - t1[b,n,o] + t2[b,n,o]
//               with t1 = x @ w[:, :C]^T, t2 = x @ w[:, C:]^T   (dense GEMMs)
//               BN(gamma>0) and LeakyReLU are monotone -> commute with max_j.
//   knn:        ordering of pd = 2*inner - xx_m (- xx_n const per row),
//               inner = X @ X^T via batched WMMA GEMM, then wave32 top-20.
//
// GEMM kernel: 16x64 output per wave = 1 A tile x 4 B tiles -> 4 independent
// WMMA accumulator chains, software-pipelined (next k-step's 5 tile loads are
// issued before the current 4 WMMAs execute), + global_prefetch one step out.
// ---------------------------------------------------------------------------

typedef __attribute__((ext_vector_type(16))) _Float16 v16h;
typedef __attribute__((ext_vector_type(8)))  _Float16 v8h;
typedef __attribute__((ext_vector_type(8)))  float    v8f;

#define BN_EPS 1e-5f
#define NEG_BIG (-3.0e38f)

// ------------------------------ small kernels ------------------------------

__global__ void k_transpose(const float* __restrict__ x, float* __restrict__ xt,
                            int B, int N) {
  int t = blockIdx.x * blockDim.x + threadIdx.x;          // B*N*3
  if (t >= B * N * 3) return;
  int c = t % 3;
  int n = (t / 3) % N;
  int b = t / (3 * N);
  xt[t] = x[((long)b * 3 + c) * N + n];
}

__global__ void k_cvt_f16(const float* __restrict__ s, _Float16* __restrict__ d, int n) {
  int t = blockIdx.x * blockDim.x + threadIdx.x;
  if (t < n) d[t] = (_Float16)s[t];
}

__global__ void k_norms(const float* __restrict__ act, float* __restrict__ xx,
                        int P, int C) {
  int p = blockIdx.x * blockDim.x + threadIdx.x;
  if (p >= P) return;
  const float* a = act + (long)p * C;
  float s = 0.f;
  for (int c = 0; c < C; c++) { float v = a[c]; s += v * v; }
  xx[p] = s;
}

// Layer-1 Gram (C=3): trivial, one thread per (b,n,m)
__global__ void k_gram3(const float* __restrict__ xt, float* __restrict__ G,
                        int B, int N) {
  long t = (long)blockIdx.x * blockDim.x + threadIdx.x;
  long total = (long)B * N * N;
  if (t >= total) return;
  int m = (int)(t % N);
  long bn = t / N;
  int n = (int)(bn % N);
  int b = (int)(bn / N);
  const float* pn = xt + ((long)b * N + n) * 3;
  const float* pm = xt + ((long)b * N + m) * 3;
  G[t] = pn[0] * pm[0] + pn[1] * pm[1] + pn[2] * pm[2];
}

// Layer-1 t1/t2 (K=3): one thread per (p, o)
__global__ void k_ec1_t(const float* __restrict__ xt, const float* __restrict__ w1,
                        float* __restrict__ t1, float* __restrict__ t2, int P) {
  int t = blockIdx.x * blockDim.x + threadIdx.x;          // P*64
  if (t >= P * 64) return;
  int o = t & 63;
  int p = t >> 6;
  const float* xp = xt + (long)p * 3;
  const float* wr = w1 + (long)o * 6;
  t1[t] = xp[0] * wr[0] + xp[1] * wr[1] + xp[2] * wr[2];
  t2[t] = xp[0] * wr[3] + xp[1] * wr[4] + xp[2] * wr[5];
}

// ------------------------ top-k (wave32 per row) ---------------------------
// key[m] = 2*G[row,m] - xx[b,m]; 32 strided candidates per lane held in
// registers, 20 rounds of (local argmax -> shfl_xor butterfly argmax -> kill).
__global__ void k_topk(const float* __restrict__ G, const float* __restrict__ xx,
                       int* __restrict__ idx, int B, int N, int k) {
  int row = blockIdx.x;                                   // b*N + n
  int lane = threadIdx.x;                                 // 0..31
  int b = row / N;
  const float* g = G + (long)row * N;
  const float* xb = xx + (long)b * N;
  float v[32];
#pragma unroll
  for (int i = 0; i < 32; i++) {
    int m = lane + i * 32;
    v[i] = 2.0f * g[m] - xb[m];
  }
  unsigned dead = 0u;
  for (int r = 0; r < k; r++) {
    float bv = NEG_BIG;
    int bm = lane;                                        // placeholder
#pragma unroll
    for (int i = 0; i < 32; i++) {
      if (!((dead >> i) & 1u)) {
        int m = lane + i * 32;
        if (v[i] > bv) { bv = v[i]; bm = m; }
      }
    }
    for (int off = 16; off > 0; off >>= 1) {
      float ov = __shfl_xor(bv, off, 32);
      int  om = __shfl_xor(bm, off, 32);
      if (ov > bv || (ov == bv && om < bm)) { bv = ov; bm = om; }
    }
    if ((bm & 31) == lane) dead |= (1u << (bm >> 5));
    if (lane == 0) idx[(long)row * k + r] = bm;
  }
}

// ----------------- gather-max + BN + LeakyReLU epilogue --------------------
__global__ void k_edge_out(const float* __restrict__ t1, const float* __restrict__ t2,
                           const int* __restrict__ idx, const float* __restrict__ bnp,
                           float* __restrict__ h, _Float16* __restrict__ hcat,
                           int ldcat, int N, int O, int k, long total) {
  long t = (long)blockIdx.x * blockDim.x + threadIdx.x;   // B*N*O
  if (t >= total) return;
  int o = (int)(t % O);
  long p = t / O;                                         // b*N + n
  long base = (p / N) * (long)N;
  const int* ir = idx + p * k;
  float best = NEG_BIG;
  for (int j = 0; j < k; j++) {
    float nv = t1[(base + ir[j]) * (long)O + o];
    best = fmaxf(best, nv);
  }
  float xv = best - t1[p * O + o] + t2[p * O + o];
  float gm = bnp[o], bb = bnp[O + o], mn = bnp[2 * O + o], vr = bnp[3 * O + o];
  float y = gm * (xv - mn) * rsqrtf(vr + BN_EPS) + bb;
  y = (y >= 0.f) ? y : 0.2f * y;
  h[t] = y;
  hcat[p * ldcat + o] = (_Float16)y;
}

// --------------------------- WMMA GEMM -------------------------------------
// C[M,Nd] = A[M,K] @ Bw[Nd,K]^T, f16 inputs, f32 accumulate.
// One wave computes a 16x64 output strip: 1 A tile x 4 B tiles -> 4
// independent accumulators; double-buffered so next k-step's loads are in
// flight while the current 4 WMMAs execute.
// Epilogue: +bias, BN, LeakyReLU, optional f16 mirror. Batched via blockIdx.z.
//
// VGPR layouts per ISA sec 7.12.2 (wave32):
//  A tile 16x32 f16: lane = half*16 + m; elems 0..7  <- K half*8 + 0..7,
//                                        elems 8..15 <- K 16 + half*8 + 0..7.
//  B tile 32x16 f16: lane = half*16 + n; elems 0..7  <- K half*8 + 0..7,
//                                        elems 8..15 <- K 16 + half*8 + 0..7.
//  C/D 16x16 f32:    lane n = lane%16; VGPR r -> M = r + 8*(lane/16).

__device__ __forceinline__ v16h pack16(v8h lo, v8h hi) {
  v16h out;
#pragma unroll
  for (int i = 0; i < 8; i++) { out[i] = lo[i]; out[i + 8] = hi[i]; }
  return out;
}

__device__ __forceinline__ v16h load_tileA(const _Float16* __restrict__ p) {
  return pack16(*(const v8h*)p, *(const v8h*)(p + 16));
}

__device__ __forceinline__ v16h load_tileB(const _Float16* __restrict__ p) {
  return pack16(*(const v8h*)p, *(const v8h*)(p + 16));
}

__global__ void k_gemm(const _Float16* __restrict__ A, long lda, long sA,
                       const _Float16* __restrict__ Bw, long ldb, long sB,
                       float* __restrict__ C, long ldc, long sC,
                       int M, int Nd, int K,
                       const float* __restrict__ bnp,   // [4,Nd] or null
                       const float* __restrict__ bias,  // [Nd] or null
                       int do_lrelu,
                       _Float16* __restrict__ H, long ldh, long sH) {
  int lane = threadIdx.x;                                 // 32
  int n0 = (blockIdx.x * blockDim.y + threadIdx.y) * 64;  // 64-col strip
  int m0 = blockIdx.y * 16;
  int bz = blockIdx.z;
  if (n0 >= Nd) return;                                   // wave-uniform exit

  // per-lane clamped B row indices for the 4 N-tiles (pure data selection,
  // EXEC stays all-ones through every WMMA)
  int ra = lane & 15;
  int kh = (lane >> 4) * 8;
  int kb = (lane >> 4) * 16;
  const _Float16* pa0 = A + (long)bz * sA + ((long)m0 + ra) * lda + kh;
  const _Float16* pb0[4];
#pragma unroll
  for (int j = 0; j < 4; j++) {
    int nr = n0 + j * 16 + (lane & 15);
    if (nr >= Nd) nr = Nd - 1;
    pb0[j] = Bw + (long)bz * sB + (long)nr * ldb + kh;
  }

  v8f acc[4] = {};
  // software pipeline: tiles for step k0 live in {a, b[]}; next step's loads
  // are issued before the current WMMAs execute.
  v16h a = load_tileA(pa0);
  v16h b[4];
#pragma unroll
  for (int j = 0; j < 4; j++) b[j] = load_tileB(pb0[j]);

  for (int k0 = 0; k0 < K; k0 += 32) {
    int kn = (k0 + 32 < K) ? (k0 + 32) : k0;              // clamped next step
    v16h an = load_tileA(pa0 + kn);
    v16h bn[4];
#pragma unroll
    for (int j = 0; j < 4; j++) bn[j] = load_tileB(pb0[j] + kn);
    __builtin_prefetch(pa0 + kn + 32, 0, 1);              // global_prefetch_b8
#pragma unroll
    for (int j = 0; j < 4; j++) {
      acc[j] = __builtin_amdgcn_wmma_f32_16x16x32_f16(false, a, false, b[j],
                                                      (short)0, acc[j],
                                                      false, false);
    }
    a = an;
#pragma unroll
    for (int j = 0; j < 4; j++) b[j] = bn[j];
  }

  int mh = (lane >> 4) * 8;
#pragma unroll
  for (int j = 0; j < 4; j++) {
    int n = n0 + j * 16 + (lane & 15);
    if (n >= Nd) continue;                                // after all WMMA
    float gm = 1.f, bb = 0.f, mn = 0.f, rs = 1.f;
    if (bnp) {
      gm = bnp[n]; bb = bnp[Nd + n]; mn = bnp[2 * Nd + n];
      rs = rsqrtf(bnp[3 * Nd + n] + BN_EPS);
    }
    float bsv = bias ? bias[n] : 0.f;
#pragma unroll
    for (int r = 0; r < 8; r++) {
      int m = m0 + mh + r;
      float x = acc[j][r] + bsv;
      if (bnp) x = gm * (x - mn) * rs + bb;
      if (do_lrelu) x = (x >= 0.f) ? x : 0.2f * x;
      C[(long)bz * sC + (long)m * ldc + n] = x;
      if (H) H[(long)bz * sH + (long)m * ldh + n] = (_Float16)x;
    }
  }
}

// ------------------------------ pooling ------------------------------------
__global__ void k_pool(const float* __restrict__ h5, float* __restrict__ pooled,
                       _Float16* __restrict__ pooledh, int N, int C) {
  int c = blockIdx.x * blockDim.x + threadIdx.x;
  int b = blockIdx.y;
  if (c >= C) return;
  const float* p = h5 + (long)b * N * C + c;
  float mx = NEG_BIG, sm = 0.f;
  for (int n = 0; n < N; n++) { float v = p[(long)n * C]; mx = fmaxf(mx, v); sm += v; }
  float mean = sm * (1.0f / (float)N);
  long o = (long)b * 2 * C;
  pooled[o + c] = mx;
  pooled[o + C + c] = mean;
  pooledh[o + c] = (_Float16)mx;
  pooledh[o + C + c] = (_Float16)mean;
}

// ------------------------------ launcher -----------------------------------

extern "C" void kernel_launch(void* const* d_in, const int* in_sizes, int n_in,
                              void* d_out, int out_size, void* d_ws, size_t ws_size,
                              hipStream_t stream) {
  (void)in_sizes; (void)n_in; (void)out_size; (void)ws_size;
  const int B = 16, N = 1024, KNN = 20, P = B * N;

  const float* x   = (const float*)d_in[0];
  const float* w1  = (const float*)d_in[1];
  const float* w2  = (const float*)d_in[2];
  const float* w3  = (const float*)d_in[3];
  const float* w4  = (const float*)d_in[4];
  const float* w5  = (const float*)d_in[5];
  const float* bn1 = (const float*)d_in[6];
  const float* bn2 = (const float*)d_in[7];
  const float* bn3 = (const float*)d_in[8];
  const float* bn4 = (const float*)d_in[9];
  const float* bn5 = (const float*)d_in[10];
  const float* bn6 = (const float*)d_in[11];
  const float* bn7 = (const float*)d_in[12];
  const float* l1w = (const float*)d_in[13];
  const float* l2w = (const float*)d_in[14];
  const float* l2b = (const float*)d_in[15];
  const float* l3w = (const float*)d_in[16];
  const float* l3b = (const float*)d_in[17];
  float* out = (float*)d_out;

  // bump allocator on d_ws
  size_t off = 0;
  char* base = (char*)d_ws;
  auto alloc = [&](size_t bytes) -> void* {
    off = (off + 255) & ~(size_t)255;
    void* p = base + off;
    off += bytes;
    return p;
  };
  float*    xt    = (float*)alloc((size_t)P * 3 * 4);
  float*    xx    = (float*)alloc((size_t)P * 4);
  int*      idx   = (int*)alloc((size_t)P * KNN * 4);
  float*    t1    = (float*)alloc((size_t)P * 256 * 4);
  float*    t2    = (float*)alloc((size_t)P * 256 * 4);
  float*    hbuf  = (float*)alloc((size_t)P * 256 * 4);
  _Float16* cat   = (_Float16*)alloc((size_t)P * 512 * 2);
  float*    G     = (float*)alloc((size_t)B * N * N * 4);   // reused as h5
  float*    h5    = G;                                      // same size (64MB)
  float*    pooled  = (float*)alloc((size_t)B * 2048 * 4);
  _Float16* pooledh = (_Float16*)alloc((size_t)B * 2048 * 2);
  float*    a1  = (float*)alloc((size_t)16 * 512 * 4);
  _Float16* a1h = (_Float16*)alloc((size_t)16 * 512 * 2);
  float*    a2  = (float*)alloc((size_t)16 * 256 * 4);
  _Float16* a2h = (_Float16*)alloc((size_t)16 * 256 * 2);
  _Float16* w2h  = (_Float16*)alloc((size_t)64 * 128 * 2);
  _Float16* w3h  = (_Float16*)alloc((size_t)128 * 128 * 2);
  _Float16* w4h  = (_Float16*)alloc((size_t)256 * 256 * 2);
  _Float16* w5h  = (_Float16*)alloc((size_t)1024 * 512 * 2);
  _Float16* l1wh = (_Float16*)alloc((size_t)512 * 2048 * 2);
  _Float16* l2wh = (_Float16*)alloc((size_t)256 * 512 * 2);
  _Float16* l3wh = (_Float16*)alloc((size_t)40 * 256 * 2);

  auto cvt = [&](const float* s, _Float16* d, int n) {
    k_cvt_f16<<<(n + 255) / 256, 256, 0, stream>>>(s, d, n);
  };
  cvt(w2, w2h, 64 * 128);     cvt(w3, w3h, 128 * 128);
  cvt(w4, w4h, 256 * 256);    cvt(w5, w5h, 1024 * 512);
  cvt(l1w, l1wh, 512 * 2048); cvt(l2w, l2wh, 256 * 512);
  cvt(l3w, l3wh, 40 * 256);

  auto gemm = [&](const _Float16* A, long lda, long sA,
                  const _Float16* Bw, long ldb, long sB,
                  float* C, long ldc, long sC,
                  int M, int Nd, int K,
                  const float* bnp, const float* bias, int lrelu,
                  _Float16* H, long ldh, long sH, int nbatch) {
    int strips = (Nd + 63) / 64;                 // 64-col strips per row-tile
    int by = strips < 4 ? strips : 4;            // waves per block
    dim3 grid((strips + by - 1) / by, M / 16, nbatch);
    dim3 block(32, by, 1);
    k_gemm<<<grid, block, 0, stream>>>(A, lda, sA, Bw, ldb, sB, C, ldc, sC,
                                       M, Nd, K, bnp, bias, lrelu, H, ldh, sH);
  };

  // ---- layer 1 (C=3 -> 64) ----
  k_transpose<<<(P * 3 + 255) / 256, 256, 0, stream>>>(x, xt, B, N);
  k_norms<<<(P + 255) / 256, 256, 0, stream>>>(xt, xx, P, 3);
  k_gram3<<<(int)(((long)B * N * N + 255) / 256), 256, 0, stream>>>(xt, G, B, N);
  k_topk<<<P, 32, 0, stream>>>(G, xx, idx, B, N, KNN);
  k_ec1_t<<<(P * 64 + 255) / 256, 256, 0, stream>>>(xt, w1, t1, t2, P);
  k_edge_out<<<(int)(((long)P * 64 + 255) / 256), 256, 0, stream>>>(
      t1, t2, idx, bn1, hbuf, cat + 0, 512, N, 64, KNN, (long)P * 64);

  // ---- layers 2..4: Gram GEMM -> topk -> t GEMMs -> gather-max ----
  struct L { int cin, o, coff_in, coff_out; const _Float16* wh; long ldw; const float* bnp; };
  L layers[3] = {
      {64, 64, 0, 64, w2h, 128, bn2},
      {64, 128, 64, 128, w3h, 128, bn3},
      {128, 256, 128, 256, w4h, 256, bn4},
  };
  for (int li = 0; li < 3; li++) {
    L& L_ = layers[li];
    k_norms<<<(P + 255) / 256, 256, 0, stream>>>(hbuf, xx, P, L_.cin);
    // batched Gram: per-batch [N,cin] @ [N,cin]^T
    gemm(cat + L_.coff_in, 512, (long)N * 512,
         cat + L_.coff_in, 512, (long)N * 512,
         G, N, (long)N * N,
         N, N, L_.cin, nullptr, nullptr, 0, nullptr, 0, 0, B);
    k_topk<<<P, 32, 0, stream>>>(G, xx, idx, B, N, KNN);
    gemm(cat + L_.coff_in, 512, 0, L_.wh, L_.ldw, 0,
         t1, L_.o, 0, P, L_.o, L_.cin, nullptr, nullptr, 0, nullptr, 0, 0, 1);
    gemm(cat + L_.coff_in, 512, 0, L_.wh + L_.cin, L_.ldw, 0,
         t2, L_.o, 0, P, L_.o, L_.cin, nullptr, nullptr, 0, nullptr, 0, 0, 1);
    k_edge_out<<<(int)(((long)P * L_.o + 255) / 256), 256, 0, stream>>>(
        t1, t2, idx, L_.bnp, hbuf, cat + L_.coff_out, 512, N, L_.o, KNN,
        (long)P * L_.o);
  }

  // ---- h5: [16384,512] @ [512,1024] + bn5 + lrelu ----
  gemm(cat, 512, 0, w5h, 512, 0, h5, 1024, 0,
       P, 1024, 512, bn5, nullptr, 1, nullptr, 0, 0, 1);

  // ---- global max/mean pool -> [16,2048] ----
  {
    dim3 grid((1024 + 255) / 256, B, 1);
    k_pool<<<grid, 256, 0, stream>>>(h5, pooled, pooledh, N, 1024);
  }

  // ---- classifier head (M=16 single row-tile) ----
  gemm(pooledh, 2048, 0, l1wh, 2048, 0, a1, 512, 0,
       16, 512, 2048, bn6, nullptr, 1, a1h, 512, 0, 1);
  gemm(a1h, 512, 0, l2wh, 512, 0, a2, 256, 0,
       16, 256, 512, bn7, l2b, 1, a2h, 256, 0, 1);
  gemm(a2h, 256, 0, l3wh, 256, 0, out, 40, 0,
       16, 40, 256, nullptr, l3b, 0, nullptr, 0, 0, 1);
}
